// QTransformer_26517128085990
// MI455X (gfx1250) — compile-verified, matrix operand order checked by reading, not run on hardware
//
#include <hip/hip_runtime.h>
#include <cstdint>
#include <cstddef>

// ---------------------------------------------------------------------------
// Types for CDNA5 WMMA (gfx1250, wave32): 16x16x32 bf16 -> f32
// ---------------------------------------------------------------------------
typedef __attribute__((ext_vector_type(16))) __bf16 v16bf;
typedef __attribute__((ext_vector_type(8)))  float  v8f;

union Frag32 { uint4 u[2]; v16bf v; };

__device__ __forceinline__ unsigned short f2bf(float f) {
  unsigned int u = __float_as_uint(f);
  u += 0x7FFFu + ((u >> 16) & 1u);          // round-to-nearest-even
  return (unsigned short)(u >> 16);
}
__device__ __forceinline__ float bf2f(unsigned short h) {
  return __uint_as_float(((unsigned int)h) << 16);
}

// Async global->LDS copy (ASYNCcnt path) if the toolchain exposes it.
#if defined(__has_builtin)
#if __has_builtin(__builtin_amdgcn_global_load_async_to_lds_b128) && \
    __has_builtin(__builtin_amdgcn_s_wait_asynccnt)
#define USE_ASYNC_COPY 1
#endif
#endif
#ifndef USE_ASYNC_COPY
#define USE_ASYNC_COPY 0
#endif

#if USE_ASYNC_COPY
// The builtin takes pointers to GCC-vector int4 (per hipcc diagnostic):
//   param0: global source, param1: LDS destination
typedef int v4i_vs __attribute__((vector_size(16)));
typedef __attribute__((address_space(1))) v4i_vs* as1_v4i;
typedef __attribute__((address_space(3))) v4i_vs* as3_v4i;
#endif

// ---------------------------------------------------------------------------
// Model constants
// ---------------------------------------------------------------------------
#define BB    1024
#define SEQL  32
#define ADIM  8
#define TT    39          // ACTION_DIM + SEQ_LEN - 1
#define HID   1024
#define NHEAD 16
#define HDIM  64
#define INTER 4096
#define NBINS 256
#define ROWS  (BB*TT)     // 39936, divisible by 128

// ---------------------------------------------------------------------------
// WMMA GEMM: C[M,N] = A[M,K] @ W[K,N],  A bf16 row-major (ld=K),
// W passed pre-transposed as Bt[N,K] bf16 row-major (ld=K).
// Block tile 128x128, BK=32 (one wmma K-step), 256 threads = 8 waves,
// wave tile 32x64 = 2x4 wmma accumulators.
// 4-deep LDS pipeline, issue-ahead 2, K-loop unrolled by 4 so all buffer
// indices are literals (no runtime modulo, branch-free steady state).
// Requires K/BK to be a multiple of 4 and >= 4 (holds: K in {1024,4096}).
// EP: 0=store bf16, 1=silu->bf16, 2=outb*=acc (bf16), 3=outf+=acc, 4=outf=acc+bias
// ---------------------------------------------------------------------------
#define BM 128
#define BN 128
#define BK 32
#define LDSS 40   // padded LDS row stride (halves)
#define NBUF 4    // pipeline depth (power of two)

template <int EP>
__global__ __launch_bounds__(256)
void gemm_wmma(const unsigned short* __restrict__ A,
               const unsigned short* __restrict__ Bt,
               unsigned short* __restrict__ outb,
               float* __restrict__ outf,
               const float* __restrict__ bias,
               int N, int K)
{
  __shared__ unsigned short As[NBUF][BM][LDSS];
  __shared__ unsigned short Bs[NBUF][BN][LDSS];

  const int tid   = threadIdx.x;
  const int lane  = tid & 31;
  const int wave  = tid >> 5;
  const int waveM = wave & 3;          // 4 waves along M
  const int waveN = wave >> 2;         // 2 waves along N
  const int lm    = lane & 15;
  const int kh    = (lane >> 4) << 3;  // 0 or 8

  const int mBase = blockIdx.y * BM;
  const int nBase = blockIdx.x * BN;

  // per-thread tile-copy coordinates: 2 x 16B for A, 2 x 16B for B
  const int rowA0 = tid >> 2;           // 0..63
  const int rowA1 = 64 + (tid >> 2);    // 64..127
  const int cvv   = (tid & 3) << 3;     // 0,8,16,24 (halves)

  // issue one BK tile (global -> LDS buffer `buf`): 4 async (or sync) 16B ops
  auto issue_tile = [&](int k0, int buf) {
#if USE_ASYNC_COPY
    __builtin_amdgcn_global_load_async_to_lds_b128(
        (as1_v4i)(A + (size_t)(mBase + rowA0) * K + k0 + cvv),
        (as3_v4i)&As[buf][rowA0][cvv], 0, 0);
    __builtin_amdgcn_global_load_async_to_lds_b128(
        (as1_v4i)(Bt + (size_t)(nBase + rowA0) * K + k0 + cvv),
        (as3_v4i)&Bs[buf][rowA0][cvv], 0, 0);
    __builtin_amdgcn_global_load_async_to_lds_b128(
        (as1_v4i)(A + (size_t)(mBase + rowA1) * K + k0 + cvv),
        (as3_v4i)&As[buf][rowA1][cvv], 0, 0);
    __builtin_amdgcn_global_load_async_to_lds_b128(
        (as1_v4i)(Bt + (size_t)(nBase + rowA1) * K + k0 + cvv),
        (as3_v4i)&Bs[buf][rowA1][cvv], 0, 0);
#else
    *(uint4*)&As[buf][rowA0][cvv] = *(const uint4*)(A  + (size_t)(mBase + rowA0) * K + k0 + cvv);
    *(uint4*)&Bs[buf][rowA0][cvv] = *(const uint4*)(Bt + (size_t)(nBase + rowA0) * K + k0 + cvv);
    *(uint4*)&As[buf][rowA1][cvv] = *(const uint4*)(A  + (size_t)(mBase + rowA1) * K + k0 + cvv);
    *(uint4*)&Bs[buf][rowA1][cvv] = *(const uint4*)(Bt + (size_t)(nBase + rowA1) * K + k0 + cvv);
#endif
  };

  v8f acc[2][4];
#pragma unroll
  for (int i = 0; i < 2; ++i)
#pragma unroll
    for (int j = 0; j < 4; ++j)
#pragma unroll
      for (int e = 0; e < 8; ++e) acc[i][j][e] = 0.0f;

  // one pipeline compute step: fragments from LDS buffer `buf`, 8 wmma
  auto compute = [&](int buf) {
    Frag32 fa[2], fb[4];
#pragma unroll
    for (int ms = 0; ms < 2; ++ms) {
      int r = waveM * 32 + ms * 16 + lm;
      fa[ms].u[0] = *(const uint4*)&As[buf][r][kh];
      fa[ms].u[1] = *(const uint4*)&As[buf][r][kh + 16];
    }
#pragma unroll
    for (int ns = 0; ns < 4; ++ns) {
      int r = waveN * 64 + ns * 16 + lm;
      fb[ns].u[0] = *(const uint4*)&Bs[buf][r][kh];
      fb[ns].u[1] = *(const uint4*)&Bs[buf][r][kh + 16];
    }
#pragma unroll
    for (int ms = 0; ms < 2; ++ms)
#pragma unroll
      for (int ns = 0; ns < 4; ++ns)
        acc[ms][ns] = __builtin_amdgcn_wmma_f32_16x16x32_bf16(
            false, fa[ms].v, false, fb[ns].v, (short)0, acc[ms][ns], false, false);
  };

  const int kt = K / BK;   // multiple of NBUF, >= NBUF

  // prologue: two tiles in flight
  issue_tile(0, 0);
  issue_tile(BK, 1);

  int k = 0;
  // main loop: branch-free steady state, all buffer indices literal
  for (; k + NBUF < kt; k += NBUF) {
#pragma unroll
    for (int j = 0; j < NBUF; ++j) {
      const int kk = k + j;
      // prefetch tile kk+3 (in range: kk <= kt-5 here)
      {
        int pr = tid & 127;
        const unsigned short* pp =
            (tid < 128) ? (A  + (size_t)(mBase + pr) * K + (kk + 3) * BK)
                        : (Bt + (size_t)(nBase + pr) * K + (kk + 3) * BK);
        __builtin_prefetch(pp, 0, 3);
      }
      issue_tile((kk + 2) * BK, (j + 2) & (NBUF - 1));
#if USE_ASYNC_COPY
      __builtin_amdgcn_s_wait_asynccnt(8);   // batch kk retired; kk+1,kk+2 in flight
#endif
      __syncthreads();                       // tile kk visible to all waves
      compute(j);
    }
  }

  // epilogue: final NBUF iterations (kk = kt-NBUF .. kt-1)
#pragma unroll
  for (int j = 0; j < NBUF; ++j) {
    const int kk = k + j;
    if (j < NBUF - 2) {
      issue_tile((kk + 2) * BK, (j + 2) & (NBUF - 1));
#if USE_ASYNC_COPY
      __builtin_amdgcn_s_wait_asynccnt(8);
#endif
    } else if (j == NBUF - 2) {
#if USE_ASYNC_COPY
      __builtin_amdgcn_s_wait_asynccnt(4);
#endif
    } else {
#if USE_ASYNC_COPY
      __builtin_amdgcn_s_wait_asynccnt(0);
#endif
    }
    __syncthreads();
    compute(j);
  }

  // Epilogue: C layout lane 0-15 -> M=i, lane 16-31 -> M=i+8 ; N = lane&15
#pragma unroll
  for (int ms = 0; ms < 2; ++ms)
#pragma unroll
    for (int ns = 0; ns < 4; ++ns)
#pragma unroll
      for (int i = 0; i < 8; ++i) {
        int m = mBase + waveM * 32 + ms * 16 + ((lane >> 4) << 3) + i;
        int n = nBase + waveN * 64 + ns * 16 + lm;
        size_t idx = (size_t)m * N + n;
        float v = acc[ms][ns][i];
        if (EP == 0) {
          outb[idx] = f2bf(v);
        } else if (EP == 1) {
          float s = v / (1.0f + __expf(-v));       // silu
          outb[idx] = f2bf(s);
        } else if (EP == 2) {
          outb[idx] = f2bf(bf2f(outb[idx]) * v);   // silu(g) * u
        } else if (EP == 3) {
          outf[idx] += v;                          // residual add
        } else {
          outf[idx] = v + bias[n];                 // head bias
        }
      }
}

// ---------------------------------------------------------------------------
// fp32 (K,N) -> bf16 transposed (N,K); blockIdx.z selects layer slice
// ---------------------------------------------------------------------------
__global__ __launch_bounds__(256)
void transpose_to_bf16(const float* __restrict__ W, unsigned short* __restrict__ Wt,
                       int K, int N)
{
  __shared__ float tile[32][33];
  const size_t mat = (size_t)blockIdx.z * K * N;
  const int n0 = blockIdx.x * 32;
  const int k0 = blockIdx.y * 32;
  for (int j = threadIdx.y; j < 32; j += 8)
    tile[j][threadIdx.x] = W[mat + (size_t)(k0 + j) * N + n0 + threadIdx.x];
  __syncthreads();
  for (int j = threadIdx.y; j < 32; j += 8)
    Wt[mat + (size_t)(n0 + j) * K + k0 + threadIdx.x] = f2bf(tile[threadIdx.x][j]);
}

// ---------------------------------------------------------------------------
// Token embedding: x[b,t,:] = states@W_state + b_state + emb_time[ts]  (t<32)
//                           = emb_action[actions[b, t-32]]             (t>=32)
// ---------------------------------------------------------------------------
__global__ __launch_bounds__(256)
void embed_kernel(const float* __restrict__ states, const int* __restrict__ actions,
                  const int* __restrict__ timesteps, const float* __restrict__ W_state,
                  const float* __restrict__ b_state, const float* __restrict__ emb_action,
                  const float* __restrict__ emb_time, float* __restrict__ x)
{
  const int row = blockIdx.x;          // b*39 + t
  const int b = row / TT;
  const int t = row % TT;
  const int tid = threadIdx.x;
  float* xr = x + (size_t)row * HID;
  if (t < SEQL) {
    __shared__ float srow[128];
    if (tid < 128) srow[tid] = states[((size_t)b * SEQL + t) * 128 + tid];
    __syncthreads();
    const int ts = timesteps[b * SEQL + t];
#pragma unroll
    for (int j = 0; j < 4; ++j) {
      int n = tid + j * 256;
      float acc = b_state[n] + emb_time[(size_t)ts * HID + n];
      for (int kk = 0; kk < 128; ++kk)
        acc += srow[kk] * W_state[(size_t)kk * HID + n];
      xr[n] = acc;
    }
  } else {
    const int a = actions[b * ADIM + (t - SEQL)];
#pragma unroll
    for (int j = 0; j < 4; ++j) {
      int n = tid + j * 256;
      xr[n] = emb_action[(size_t)a * HID + n];
    }
  }
}

// ---------------------------------------------------------------------------
// RMSNorm: h[row] = bf16( x[row] * rsqrt(mean(x^2)+eps) * w )
// ---------------------------------------------------------------------------
__global__ __launch_bounds__(256)
void rmsnorm_kernel(const float* __restrict__ x, const float* __restrict__ w,
                    unsigned short* __restrict__ h)
{
  __shared__ float red[256];
  const int row = blockIdx.x;
  const int tid = threadIdx.x;
  const float* xr = x + (size_t)row * HID;
  float v[4]; float s = 0.0f;
#pragma unroll
  for (int j = 0; j < 4; ++j) { v[j] = xr[tid + j * 256]; s += v[j] * v[j]; }
  red[tid] = s; __syncthreads();
  for (int o = 128; o > 0; o >>= 1) { if (tid < o) red[tid] += red[tid + o]; __syncthreads(); }
  const float rs = rsqrtf(red[0] * (1.0f / 1024.0f) + 1e-6f);
#pragma unroll
  for (int j = 0; j < 4; ++j) {
    int n = tid + j * 256;
    h[(size_t)row * HID + n] = f2bf(v[j] * rs * w[n]);
  }
}

// Tail RMSNorm: gather last 8 tokens per batch into compact hf[B*8, HID]
__global__ __launch_bounds__(256)
void rmsnorm_tail_kernel(const float* __restrict__ x, const float* __restrict__ w,
                         unsigned short* __restrict__ hf)
{
  __shared__ float red[256];
  const int orow = blockIdx.x;                 // b*8 + j
  const int b = orow >> 3;
  const int t = (TT - ADIM) + (orow & 7);      // 31..38
  const int row = b * TT + t;
  const int tid = threadIdx.x;
  const float* xr = x + (size_t)row * HID;
  float v[4]; float s = 0.0f;
#pragma unroll
  for (int j = 0; j < 4; ++j) { v[j] = xr[tid + j * 256]; s += v[j] * v[j]; }
  red[tid] = s; __syncthreads();
  for (int o = 128; o > 0; o >>= 1) { if (tid < o) red[tid] += red[tid + o]; __syncthreads(); }
  const float rs = rsqrtf(red[0] * (1.0f / 1024.0f) + 1e-6f);
#pragma unroll
  for (int j = 0; j < 4; ++j) {
    int n = tid + j * 256;
    hf[(size_t)orow * HID + n] = f2bf(v[j] * rs * w[n]);
  }
}

// ---------------------------------------------------------------------------
// RoPE in-place on bf16 q and k. One thread per (row, head, d<32) pair.
// ---------------------------------------------------------------------------
__global__ __launch_bounds__(256)
void rope_kernel(unsigned short* __restrict__ q, unsigned short* __restrict__ k, int total)
{
  int i = blockIdx.x * 256 + threadIdx.x;
  if (i >= total) return;
  const int d  = i & 31;
  const int hh = (i >> 5) & 15;
  const int bt = i >> 9;               // b*T + t
  const int t  = bt % TT;
  const size_t base = ((size_t)bt * NHEAD + hh) * HDIM;
  const float inv = __expf(-(float)d * 0.28782313662425572f); // ln(10000)/32
  const float ang = (float)t * inv;
  const float c = __cosf(ang), sn = __sinf(ang);
  {
    float x1 = bf2f(q[base + d]), x2 = bf2f(q[base + d + 32]);
    q[base + d]      = f2bf(x1 * c - x2 * sn);
    q[base + d + 32] = f2bf(x2 * c + x1 * sn);
  }
  {
    float x1 = bf2f(k[base + d]), x2 = bf2f(k[base + d + 32]);
    k[base + d]      = f2bf(x1 * c - x2 * sn);
    k[base + d + 32] = f2bf(x2 * c + x1 * sn);
  }
}

// ---------------------------------------------------------------------------
// Attention: one block per (b, head). T=39, HDIM=64; all LDS, fp32 math.
// mask: allowed iff (s<=t) || (s<32 && t<32)
// ---------------------------------------------------------------------------
__global__ __launch_bounds__(256)
void attention_kernel(const unsigned short* __restrict__ q,
                      const unsigned short* __restrict__ k,
                      const unsigned short* __restrict__ v,
                      unsigned short* __restrict__ ctx)
{
  __shared__ float Qs[TT * HDIM], Ks[TT * HDIM], Vs[TT * HDIM];
  __shared__ float Sc[TT * 40];
  const int bh = blockIdx.x;
  const int b = bh >> 4;
  const int h = bh & 15;
  const size_t baseRow = (size_t)b * TT * HID + (size_t)h * HDIM;

  for (int i = threadIdx.x; i < TT * HDIM; i += 256) {
    int t = i >> 6, d = i & 63;
    size_t g = baseRow + (size_t)t * HID + d;
    Qs[i] = bf2f(q[g]); Ks[i] = bf2f(k[g]); Vs[i] = bf2f(v[g]);
  }
  __syncthreads();

  const float scale = 0.125f;   // 1/sqrt(64)
  for (int i = threadIdx.x; i < TT * TT; i += 256) {
    int t = i / TT, s = i % TT;
    bool ok = (s <= t) || (s < SEQL && t < SEQL);
    float acc = -1e30f;
    if (ok) {
      acc = 0.0f;
      for (int d = 0; d < HDIM; ++d) acc += Qs[t * HDIM + d] * Ks[s * HDIM + d];
      acc *= scale;
    }
    Sc[t * 40 + s] = acc;
  }
  __syncthreads();

  if (threadIdx.x < TT) {
    int t = threadIdx.x;
    float mx = -1e30f;
    for (int s = 0; s < TT; ++s) mx = fmaxf(mx, Sc[t * 40 + s]);
    float sum = 0.0f;
    for (int s = 0; s < TT; ++s) { float e = __expf(Sc[t * 40 + s] - mx); Sc[t * 40 + s] = e; sum += e; }
    float invs = 1.0f / sum;
    for (int s = 0; s < TT; ++s) Sc[t * 40 + s] *= invs;
  }
  __syncthreads();

  for (int i = threadIdx.x; i < TT * HDIM; i += 256) {
    int t = i >> 6, d = i & 63;
    float acc = 0.0f;
    for (int s = 0; s < TT; ++s) acc += Sc[t * 40 + s] * Vs[s * HDIM + d];
    ctx[baseRow + (size_t)t * HID + d] = f2bf(acc);
  }
}

// ---------------------------------------------------------------------------
// Host-side helpers
// ---------------------------------------------------------------------------
static void launch_gemm(int ep, const unsigned short* A, const unsigned short* Bt,
                        unsigned short* ob, float* of, const float* bias,
                        int M, int N, int K, hipStream_t s)
{
  dim3 g(N / BN, M / BM), b(256);
  switch (ep) {
    case 0: gemm_wmma<0><<<g, b, 0, s>>>(A, Bt, ob, of, bias, N, K); break;
    case 1: gemm_wmma<1><<<g, b, 0, s>>>(A, Bt, ob, of, bias, N, K); break;
    case 2: gemm_wmma<2><<<g, b, 0, s>>>(A, Bt, ob, of, bias, N, K); break;
    case 3: gemm_wmma<3><<<g, b, 0, s>>>(A, Bt, ob, of, bias, N, K); break;
    default: gemm_wmma<4><<<g, b, 0, s>>>(A, Bt, ob, of, bias, N, K); break;
  }
}

extern "C" void kernel_launch(void* const* d_in, const int* in_sizes, int n_in,
                              void* d_out, int out_size, void* d_ws, size_t ws_size,
                              hipStream_t stream)
{
  (void)in_sizes; (void)n_in; (void)out_size; (void)ws_size;

  const float* states     = (const float*)d_in[0];
  const int*   actions    = (const int*)  d_in[1];
  const int*   timesteps  = (const int*)  d_in[2];
  const float* W_state    = (const float*)d_in[3];
  const float* b_state    = (const float*)d_in[4];
  const float* emb_action = (const float*)d_in[5];
  const float* emb_time   = (const float*)d_in[6];
  const float* Wq         = (const float*)d_in[7];
  const float* Wk         = (const float*)d_in[8];
  const float* Wv         = (const float*)d_in[9];
  const float* Wo         = (const float*)d_in[10];
  const float* Wg         = (const float*)d_in[11];
  const float* Wu         = (const float*)d_in[12];
  const float* Wd         = (const float*)d_in[13];
  const float* ln1        = (const float*)d_in[14];
  const float* ln2        = (const float*)d_in[15];
  const float* lnf        = (const float*)d_in[16];
  const float* W_out      = (const float*)d_in[17];
  const float* b_out      = (const float*)d_in[18];
  float* out = (float*)d_out;

  // ---- workspace carve-up ----
  size_t off = 0;
  auto take = [&](size_t bytes) -> void* {
    void* p = (char*)d_ws + off;
    off += (bytes + 255) & ~(size_t)255;
    return p;
  };
  const size_t WH = (size_t)HID * HID;     // 1M elems per layer (attn proj)
  const size_t WI = (size_t)HID * INTER;   // 4M elems per layer (mlp)

  unsigned short* wqT   = (unsigned short*)take(6 * WH * 2);
  unsigned short* wkT   = (unsigned short*)take(6 * WH * 2);
  unsigned short* wvT   = (unsigned short*)take(6 * WH * 2);
  unsigned short* woT   = (unsigned short*)take(6 * WH * 2);
  unsigned short* wgT   = (unsigned short*)take(6 * WI * 2);
  unsigned short* wuT   = (unsigned short*)take(6 * WI * 2);
  unsigned short* wdT   = (unsigned short*)take(6 * WI * 2);
  unsigned short* woutT = (unsigned short*)take((size_t)HID * NBINS * 2);
  float*          x     = (float*)take((size_t)ROWS * HID * 4);
  unsigned short* h     = (unsigned short*)take((size_t)ROWS * HID * 2);
  // big region: q|k|v (3 * ROWS*HID bf16 = 245MB) overlaid with gu (ROWS*INTER bf16 = 327MB)
  unsigned short* big   = (unsigned short*)take((size_t)ROWS * INTER * 2);
  unsigned short* qb = big;
  unsigned short* kb = big + (size_t)ROWS * HID;
  unsigned short* vb = big + 2 * (size_t)ROWS * HID;
  unsigned short* gu = big;
  unsigned short* hf = (unsigned short*)take((size_t)BB * ADIM * HID * 2);

  // ---- weight convert + transpose (fp32 KxN -> bf16 NxK) ----
  dim3 tb(32, 8);
  transpose_to_bf16<<<dim3(HID / 32, HID / 32, 6), tb, 0, stream>>>(Wq, wqT, HID, HID);
  transpose_to_bf16<<<dim3(HID / 32, HID / 32, 6), tb, 0, stream>>>(Wk, wkT, HID, HID);
  transpose_to_bf16<<<dim3(HID / 32, HID / 32, 6), tb, 0, stream>>>(Wv, wvT, HID, HID);
  transpose_to_bf16<<<dim3(HID / 32, HID / 32, 6), tb, 0, stream>>>(Wo, woT, HID, HID);
  transpose_to_bf16<<<dim3(INTER / 32, HID / 32, 6), tb, 0, stream>>>(Wg, wgT, HID, INTER);
  transpose_to_bf16<<<dim3(INTER / 32, HID / 32, 6), tb, 0, stream>>>(Wu, wuT, HID, INTER);
  transpose_to_bf16<<<dim3(HID / 32, INTER / 32, 6), tb, 0, stream>>>(Wd, wdT, INTER, HID);
  transpose_to_bf16<<<dim3(NBINS / 32, HID / 32, 1), tb, 0, stream>>>(W_out, woutT, HID, NBINS);

  // ---- embedding ----
  embed_kernel<<<ROWS, 256, 0, stream>>>(states, actions, timesteps, W_state, b_state,
                                         emb_action, emb_time, x);

  const int ropeTotal = ROWS * NHEAD * (HDIM / 2);   // 20,447,232
  const int ropeBlocks = (ropeTotal + 255) / 256;

  // ---- transformer layers ----
  for (int l = 0; l < 6; ++l) {
    rmsnorm_kernel<<<ROWS, 256, 0, stream>>>(x, ln1 + (size_t)l * HID, h);
    launch_gemm(0, h, wqT + (size_t)l * WH, qb, nullptr, nullptr, ROWS, HID, HID, stream);
    launch_gemm(0, h, wkT + (size_t)l * WH, kb, nullptr, nullptr, ROWS, HID, HID, stream);
    launch_gemm(0, h, wvT + (size_t)l * WH, vb, nullptr, nullptr, ROWS, HID, HID, stream);
    rope_kernel<<<ropeBlocks, 256, 0, stream>>>(qb, kb, ropeTotal);
    attention_kernel<<<BB * NHEAD, 256, 0, stream>>>(qb, kb, vb, h);   // ctx -> h
    launch_gemm(3, h, woT + (size_t)l * WH, nullptr, x, nullptr, ROWS, HID, HID, stream);

    rmsnorm_kernel<<<ROWS, 256, 0, stream>>>(x, ln2 + (size_t)l * HID, h);
    launch_gemm(1, h, wgT + (size_t)l * WI, gu, nullptr, nullptr, ROWS, INTER, HID, stream); // silu(g)
    launch_gemm(2, h, wuT + (size_t)l * WI, gu, nullptr, nullptr, ROWS, INTER, HID, stream); // *= u
    launch_gemm(3, gu, wdT + (size_t)l * WI, nullptr, x, nullptr, ROWS, HID, INTER, stream); // x += down
  }

  // ---- head: rmsnorm on last 8 tokens, then @ W_out + b_out ----
  rmsnorm_tail_kernel<<<BB * ADIM, 256, 0, stream>>>(x, lnf, hf);
  launch_gemm(4, hf, woutT, nullptr, out, b_out, BB * ADIM, NBINS, HID, stream);
}
